// VQVAE_62569083568534
// MI455X (gfx1250) — matrix operand order repaired and checked by reading, not compiled, not run on hardware
//
#include <hip/hip_runtime.h>
#include <hip/hip_bf16.h>
#include <stdint.h>

// ---------------------------------------------------------------------------
// CDNA5 (gfx1250) VQVAE forward.
// GEMMs + LSTM recurrent matvecs via v_wmma_f32_16x16x32_bf16.
// GEMM: 2x4 register tiling per wave (8 f32 accumulators) + async global->LDS
// DMA staging of the shared B tile (double buffered, ASYNCcnt-tracked), so the
// 8 waves of a block fetch each B fragment from L2 exactly once.
// Gate rows remapped to stride-512 layout [i|f|g|o] so tiles align.
// ---------------------------------------------------------------------------

typedef __bf16 bf16_t;
typedef bf16_t bf16x16 __attribute__((ext_vector_type(16)));
typedef float  v8f     __attribute__((ext_vector_type(8)));

union FragU { uint4 q[2]; bf16x16 v; };

// ---------------------- async global->LDS primitives -----------------------
// One instruction moves 16B per active lane into LDS (VDST = wave-relative
// LDS byte address = low 32 bits of the generic pointer, ISA 10.2).
__device__ __forceinline__ void async_b128_to_lds(void* lds_ptr, const void* gptr) {
  uint32_t l = (uint32_t)(uintptr_t)lds_ptr;
  asm volatile("global_load_async_to_lds_b128 %0, %1, off"
               :: "v"(l), "v"(gptr) : "memory");
}
__device__ __forceinline__ void wait_async_le1() {
  asm volatile("s_wait_asynccnt 0x1" ::: "memory");
}
__device__ __forceinline__ void wait_async_0() {
  asm volatile("s_wait_asynccnt 0x0" ::: "memory");
}

// ------------------------- WMMA fragment loaders ---------------------------
// A fragment (16x32, 16-bit, row-major MxK source):
//   lane L: row = L&15, K base = kc*32 + (L>>4)*8, chunks at +0..7 and +16..23
__device__ __forceinline__ bf16x16 load_a_frag(const bf16_t* __restrict__ A,
                                               int lda, int row0, int kc, int lane) {
  FragU f;
  int r  = row0 + (lane & 15);
  int kb = kc * 32 + ((lane >> 4) << 3);
  const bf16_t* p = A + (size_t)r * lda + kb;
  f.q[0] = *(const uint4*)(p);
  f.q[1] = *(const uint4*)(p + 16);
  return f.v;
}

// B fragment (32x16, 16-bit). B = W^T with W row-major (NxK):
//   lane L: col n = L&15, K base = kc*32 + (L>>4)*8, chunks at +0..7 and +16..23
__device__ __forceinline__ bf16x16 load_b_frag(const bf16_t* __restrict__ W,
                                               int ldw, int col0, int kc, int lane) {
  FragU f;
  int n  = col0 + (lane & 15);
  int kb = kc * 32 + ((lane >> 4) << 3);
  const bf16_t* p = W + (size_t)n * ldw + kb;
  f.q[0] = *(const uint4*)(p);
  f.q[1] = *(const uint4*)(p + 16);
  return f.v;
}

// Same B fragment but from an LDS-staged tile: row stride BROW bf16 (padded
// to spread banks), tile base points at the j-th 16-row group.
#define BROW 40
__device__ __forceinline__ bf16x16 load_b_frag_lds(const bf16_t* base, int lane) {
  FragU f;
  const bf16_t* p = base + (lane & 15) * BROW + ((lane >> 4) << 3);
  f.q[0] = *(const uint4*)(p);
  f.q[1] = *(const uint4*)(p + 16);
  return f.v;
}

// A fragment holding a single vector h in row 0 (rows 1..15 zero), h in LDS.
__device__ __forceinline__ bf16x16 load_hvec_frag(const bf16_t* h, int kc, int lane) {
  FragU f;
  f.q[0] = uint4{0u, 0u, 0u, 0u};
  f.q[1] = uint4{0u, 0u, 0u, 0u};
  if ((lane & 15) == 0) {
    int kb = kc * 32 + ((lane >> 4) << 3);
    f.q[0] = *(const uint4*)(h + kb);
    f.q[1] = *(const uint4*)(h + kb + 16);
  }
  return f.v;
}

// ------------------------------ GEMM kernel --------------------------------
// out(MxN, f32, ld=ldo) = act(A(bf16, MxKp) @ W(bf16, NxKp)^T + bias)
// block = 256 (8 waves). Each wave owns a 32x64 block (2 M-tiles x 4 N-tiles).
// The 64x32 B tile shared by all waves is DMA'd into LDS (double buffered).
// grid = (N/64, M/256). M must be a multiple of 256, N of 64.
__global__ __launch_bounds__(256)
void k_wmma_gemm(const bf16_t* __restrict__ A, int lda,
                 const bf16_t* __restrict__ W, int ldw,
                 const float* __restrict__ bias,
                 float* __restrict__ out, int ldo,
                 int kchunks, int relu) {
  const int tid  = threadIdx.x;
  const int lane = tid & 31;
  const int wave = tid >> 5;
  const int row0 = (blockIdx.y * 8 + wave) * 32;   // 2 M-tiles
  const int col0 = blockIdx.x * 64;                // 4 N-tiles

  __shared__ __align__(16) bf16_t b_s[2][64 * BROW];

  // Cooperative async stage of one 64x32 B tile: 256 threads x 16B = 4KB.
  const int wrow = tid >> 2;            // W row within tile (0..63)
  const int wchk = (tid & 3) << 3;      // bf16 offset 0/8/16/24 within the row
  auto issue_tile = [&](int kc, int buf) {
    const bf16_t* g = W + (size_t)(col0 + wrow) * ldw + kc * 32 + wchk;
    async_b128_to_lds(&b_s[buf][wrow * BROW + wchk], g);
  };

  const v8f vzero = {0.f, 0.f, 0.f, 0.f, 0.f, 0.f, 0.f, 0.f};
  v8f acc[2][4];
#pragma unroll
  for (int mi = 0; mi < 2; ++mi)
#pragma unroll
    for (int j = 0; j < 4; ++j) acc[mi][j] = vzero;

  issue_tile(0, 0);
  for (int kc = 0; kc < kchunks; ++kc) {
    const int buf = kc & 1;
    if (kc + 1 < kchunks) { issue_tile(kc + 1, buf ^ 1); wait_async_le1(); }
    else                  { wait_async_0(); }
    __syncthreads();                        // staged tile visible block-wide

    bf16x16 a0 = load_a_frag(A, lda, row0,      kc, lane);
    bf16x16 a1 = load_a_frag(A, lda, row0 + 16, kc, lane);
#pragma unroll
    for (int j = 0; j < 4; ++j) {
      bf16x16 b = load_b_frag_lds(&b_s[buf][j * 16 * BROW], lane);
      acc[0][j] = __builtin_amdgcn_wmma_f32_16x16x32_bf16(false, a0, false, b,
                                                          (short)0, acc[0][j],
                                                          false, false);
      acc[1][j] = __builtin_amdgcn_wmma_f32_16x16x32_bf16(false, a1, false, b,
                                                          (short)0, acc[1][j],
                                                          false, false);
    }
    __syncthreads();                        // done reading buf before overwrite
  }

#pragma unroll
  for (int mi = 0; mi < 2; ++mi) {
    int r0 = row0 + mi * 16 + ((lane >> 4) << 3);
#pragma unroll
    for (int j = 0; j < 4; ++j) {
      int col = col0 + j * 16 + (lane & 15);
      float bv = bias ? bias[col] : 0.0f;
#pragma unroll
      for (int i = 0; i < 8; ++i) {
        float v = acc[mi][j][i] + bv;
        if (relu) v = fmaxf(v, 0.0f);
        out[(size_t)(r0 + i) * ldo + col] = v;
      }
    }
  }
}

// --------------------------- LSTM recurrence -------------------------------
// One workgroup (1024 thr = 32 waves) per direction. Padded gate layout:
// [i:0..511 | f:512.. | g:1024.. | o:1536..], padded hidden = 512.
// gbuf = precomputed x@Wih^T + (bih+bhh), T x 2048 f32.
// Whh packed bf16 2048x512 (L2-resident; each fragment read once per step by
// exactly one wave, so no LDS staging helps here).
__global__ __launch_bounds__(1024)
void k_lstm_rec(const float* __restrict__ gbuf_f, const float* __restrict__ gbuf_b,
                const bf16_t* __restrict__ whh_f, const bf16_t* __restrict__ whh_b,
                float* __restrict__ out, int ldo, int T, int hid) {
  const int dir = blockIdx.x;
  const float*  gbuf = dir ? gbuf_b : gbuf_f;
  const bf16_t* whh  = dir ? whh_b  : whh_f;

  __shared__ __align__(16) bf16_t h_s[512];
  __shared__ float c_s[512];
  __shared__ float g_s[2048];

  const int tid  = threadIdx.x;
  const int lane = tid & 31;
  const int wave = tid >> 5;

  for (int i = tid; i < 512; i += 1024) { h_s[i] = (bf16_t)0.0f; c_s[i] = 0.0f; }
  __syncthreads();

  const v8f vzero = {0.f, 0.f, 0.f, 0.f, 0.f, 0.f, 0.f, 0.f};

  for (int t = 0; t < T; ++t) {
    const int tt = dir ? (T - 1 - t) : t;
    const float* grow = gbuf + (size_t)tt * 2048;

    v8f acc[4];
#pragma unroll
    for (int j = 0; j < 4; ++j) acc[j] = vzero;

    for (int kc = 0; kc < 16; ++kc) {            // K = 512 hidden (padded)
      bf16x16 a = load_hvec_frag(h_s, kc, lane);
#pragma unroll
      for (int j = 0; j < 4; ++j) {              // 4 N-tiles per wave, 32 waves
        int tile = wave * 4 + j;
        bf16x16 b = load_b_frag(whh, 512, tile * 16, kc, lane);
        acc[j] = __builtin_amdgcn_wmma_f32_16x16x32_bf16(false, a, false, b,
                                                         (short)0, acc[j],
                                                         false, false);
      }
    }
    if (lane < 16) {                             // D row 0 = element 0, lanes 0..15
#pragma unroll
      for (int j = 0; j < 4; ++j) {
        int col = (wave * 4 + j) * 16 + lane;
        g_s[col] = acc[j][0] + grow[col];
        // warm next step's xproj row while the cell update runs
        const int ttn = dir ? (tt - 1) : (tt + 1);
        if (ttn >= 0 && ttn < T)
          __builtin_prefetch(gbuf + (size_t)ttn * 2048 + col, 0, 1);
      }
    }
    __syncthreads();

    if (tid < 512) {                             // elementwise cell update
      int n = tid;
      float gi = g_s[n], gf = g_s[512 + n], gg = g_s[1024 + n], go = g_s[1536 + n];
      float si = 1.0f / (1.0f + __expf(-gi));
      float sf = 1.0f / (1.0f + __expf(-gf));
      float tg = tanhf(gg);
      float so = 1.0f / (1.0f + __expf(-go));
      float c = sf * c_s[n] + si * tg;
      float h = so * tanhf(c);
      if (n >= hid) { c = 0.0f; h = 0.0f; }      // keep padding clean
      c_s[n] = c;
      h_s[n] = (bf16_t)h;
      if (n < hid) out[(size_t)tt * ldo + dir * hid + n] = h;
    }
    __syncthreads();
  }
}

// --------------------------- packing / builders ----------------------------
__global__ void k_pack_bf16(const float* __restrict__ src, int srows, int scols,
                            int sld, bf16_t* __restrict__ dst, int drows, int dcols) {
  size_t i = (size_t)blockIdx.x * blockDim.x + threadIdx.x;
  if (i >= (size_t)drows * dcols) return;
  int r = (int)(i / dcols), c = (int)(i % dcols);
  float v = (r < srows && c < scols) ? src[(size_t)r * sld + c] : 0.0f;
  dst[i] = (bf16_t)v;
}

// Gate-remapped pack: src (4*hid x in) -> dst (2048 x dld), gate stride 512.
__global__ void k_pack_gate_w(const float* __restrict__ W, int in_dim, int hid,
                              bf16_t* __restrict__ dst, int dld) {
  size_t i = (size_t)blockIdx.x * blockDim.x + threadIdx.x;
  if (i >= (size_t)2048 * dld) return;
  int r = (int)(i / dld), c = (int)(i % dld);
  int gate = r >> 9, u = r & 511;
  float v = (u < hid && c < in_dim) ? W[(size_t)(gate * hid + u) * in_dim + c] : 0.0f;
  dst[i] = (bf16_t)v;
}

__global__ void k_pack_gate_bias(const float* __restrict__ bih,
                                 const float* __restrict__ bhh,
                                 int hid, float* __restrict__ dst) {
  int i = blockIdx.x * blockDim.x + threadIdx.x;
  if (i >= 2048) return;
  int g = i >> 9, u = i & 511;
  dst[i] = (u < hid) ? (bih[g * hid + u] + bhh[g * hid + u]) : 0.0f;
}

__global__ void k_pack_biasf(const float* __restrict__ src, int n,
                             float* __restrict__ dst, int np) {
  int i = blockIdx.x * blockDim.x + threadIdx.x;
  if (i < np) dst[i] = (i < n) ? src[i] : 0.0f;
}

// dst(rows x dcols bf16) = [s0(:,0:c0) | s1(:,0:c1) | labels(1,0) | zeros]
__global__ void k_build_bf16(const float* __restrict__ s0, int c0, int ld0,
                             const float* __restrict__ s1, int c1, int ld1,
                             int labels, int relu,
                             bf16_t* __restrict__ dst, int rows, int dcols) {
  size_t i = (size_t)blockIdx.x * blockDim.x + threadIdx.x;
  if (i >= (size_t)rows * dcols) return;
  int r = (int)(i / dcols), c = (int)(i % dcols);
  float v = 0.0f;
  if (c < c0)            v = s0[(size_t)r * ld0 + c];
  else if (c < c0 + c1)  v = s1[(size_t)r * ld1 + (c - c0)];
  else if (labels && c == c0 + c1) v = 1.0f;
  if (relu) v = fmaxf(v, 0.0f);
  dst[i] = (bf16_t)v;
}

// fc2 input gather: rows of relu(out1) at mora / mib indices + labels.
__global__ void k_gather_fc2(const float* __restrict__ out1, int ldo,
                             const int* __restrict__ mora, int Mn, int hid, int T,
                             bf16_t* __restrict__ dst) {
  size_t i = (size_t)blockIdx.x * blockDim.x + threadIdx.x;
  if (i >= (size_t)Mn * 1024) return;
  int m = (int)(i >> 10), c = (int)(i & 1023);
  float v = 0.0f;
  if (c < hid) {
    int t = mora[m]; if (t > T - 1) t = T - 1;
    v = fmaxf(out1[(size_t)t * ldo + c], 0.0f);
  } else if (c < 2 * hid) {
    int t = (m == 0) ? 0 : (mora[m - 1] + 1); if (t > T - 1) t = T - 1;
    v = fmaxf(out1[(size_t)t * ldo + hid + (c - hid)], 0.0f);
  } else if (c == 2 * hid) {
    v = 1.0f;
  }
  dst[i] = (bf16_t)v;
}

// VQ: z = cb[argmin_k ||z_unq - cb_k||^2]  (forward value of the STE expr)
__global__ void k_vq(const float* __restrict__ zu, int ldzu,
                     const float* __restrict__ cb,
                     float* __restrict__ zbuf, float* __restrict__ out_z,
                     float* __restrict__ out_zu, int Mn) {
  int m = blockIdx.x * blockDim.x + threadIdx.x;
  if (m >= Mn) return;
  float best = 3.402823e38f; int bi = 0;
  for (int k = 0; k < 64; ++k) {
    float d = 0.0f;
#pragma unroll
    for (int j = 0; j < 16; ++j) { float t = zu[(size_t)m * ldzu + j] - cb[k * 16 + j]; d += t * t; }
    if (d < best) { best = d; bi = k; }
  }
#pragma unroll
  for (int j = 0; j < 16; ++j) {
    float zv = cb[bi * 16 + j];
    zbuf[m * 16 + j] = zv;
    out_z[m * 16 + j] = zv;
    out_zu[m * 16 + j] = zu[(size_t)m * ldzu + j];
  }
}

// z_tmp[t] = (seg < Mn) ? z[seg] : 0, seg = upper_bound(mora, t)
__global__ void k_ztmp(const int* __restrict__ mora, int Mn,
                       const float* __restrict__ z, float* __restrict__ zt, int T) {
  int t = blockIdx.x * blockDim.x + threadIdx.x;
  if (t >= T) return;
  int lo = 0, hi = Mn;
  while (lo < hi) { int mid = (lo + hi) >> 1; if (mora[mid] <= t) lo = mid + 1; else hi = mid; }
#pragma unroll
  for (int j = 0; j < 16; ++j) zt[t * 16 + j] = (lo < Mn) ? z[lo * 16 + j] : 0.0f;
}

__global__ void k_copy_y(const float* __restrict__ f, int ldf,
                         float* __restrict__ y, int T) {
  int t = blockIdx.x * blockDim.x + threadIdx.x;
  if (t < T) y[t] = f[(size_t)t * ldf];
}

// ------------------------------- host side ---------------------------------
#define CEILDIV(a, b) (((a) + (b) - 1) / (b))

extern "C" void kernel_launch(void* const* d_in, const int* in_sizes, int n_in,
                              void* d_out, int out_size, void* d_ws, size_t ws_size,
                              hipStream_t stream) {
  (void)in_sizes; (void)n_in; (void)out_size; (void)ws_size;
  const int T = 4096, HID = 511, LING = 442, AC = 199, MN = 512;

  // d_in flatten order (dict order; params pytree sorted by key):
  const float* ling   = (const float*)d_in[0];
  const float* ac     = (const float*)d_in[1];
  const float* cb     = (const float*)d_in[2];        // codebook 64x16
  const float* fc11b  = (const float*)d_in[3];
  const float* fc11w  = (const float*)d_in[4];        // 639 x 641
  const float* fc12b  = (const float*)d_in[5];
  const float* fc12w  = (const float*)d_in[6];        // 456 x 458
  const float* fc2b   = (const float*)d_in[7];
  const float* fc2w   = (const float*)d_in[8];        // 16 x 1024
  const float* fc3b   = (const float*)d_in[9];
  const float* fc3w   = (const float*)d_in[10];       // 1 x 1024
  // lstm dirs: [Whh, Wih, bhh, bih] per dir; bwd before fwd per layer.
  struct DirP { int whh, wih, bhh, bih; };
  const DirP dirs[8] = {
    {15, 16, 17, 18}, {11, 12, 13, 14},   // lstm1 L0: fwd, bwd
    {23, 24, 25, 26}, {19, 20, 21, 22},   // lstm1 L1
    {31, 32, 33, 34}, {27, 28, 29, 30},   // lstm2 L0
    {39, 40, 41, 42}, {35, 36, 37, 38},   // lstm2 L1
  };
  const int dir_in [8] = {641, 641, 1022, 1022, 458, 458, 1022, 1022};
  const int dir_kp [8] = {672, 672, 1024, 1024, 480, 480, 1024, 1024};
  const int* mora = (const int*)d_in[43];

  float* out_y  = (float*)d_out;            // 4096
  float* out_z  = out_y + T;                // 512*16
  float* out_zu = out_z + MN * 16;          // 512*16

  // ---- workspace bump allocator ----
  char*  base = (char*)d_ws;
  size_t off  = 0;
  auto alloc = [&](size_t bytes) -> void* {
    off = (off + 255) & ~(size_t)255;
    void* p = base + off; off += bytes; return p;
  };

  // N dims padded to multiples of 64 for the 2x4-tiled GEMM.
  bf16_t* wfc11 = (bf16_t*)alloc((size_t)640 * 672 * 2);
  bf16_t* wfc12 = (bf16_t*)alloc((size_t)512 * 480 * 2);
  bf16_t* wfc2  = (bf16_t*)alloc((size_t)64 * 1024 * 2);
  bf16_t* wfc3  = (bf16_t*)alloc((size_t)64 * 1024 * 2);
  float*  bfc11 = (float*)alloc(640 * 4);
  float*  bfc12 = (float*)alloc(512 * 4);
  float*  bfc2  = (float*)alloc(64 * 4);
  float*  bfc3  = (float*)alloc(64 * 4);
  bf16_t* wih_pk[8]; bf16_t* whh_pk[8]; float* gb[8];
  for (int d = 0; d < 8; ++d) {
    wih_pk[d] = (bf16_t*)alloc((size_t)2048 * dir_kp[d] * 2);
    whh_pk[d] = (bf16_t*)alloc((size_t)2048 * 512 * 2);
    gb[d]     = (float*)alloc(2048 * 4);
  }
  bf16_t* xbf   = (bf16_t*)alloc((size_t)T * 1024 * 2);    // shared input staging
  float*  fbuf  = (float*)alloc((size_t)T * 640 * 4);      // fc11/fc12/fc3 outputs
  float*  gbufF = (float*)alloc((size_t)T * 2048 * 4);
  float*  gbufB = (float*)alloc((size_t)T * 2048 * 4);
  float*  outA  = (float*)alloc((size_t)T * 1024 * 4);
  float*  outB  = (float*)alloc((size_t)T * 1024 * 4);
  float*  zubuf = (float*)alloc((size_t)MN * 64 * 4);
  float*  zbuf  = (float*)alloc((size_t)MN * 16 * 4);
  float*  ztmp  = (float*)alloc((size_t)T * 16 * 4);

  auto ew = [](size_t n) { return dim3((unsigned)CEILDIV(n, (size_t)256)); };

  // ---- phase 0: pack weights (bf16, padded, gate-remapped) ----
  k_pack_bf16<<<ew((size_t)640 * 672), 256, 0, stream>>>(fc11w, 639, 641, 641, wfc11, 640, 672);
  k_pack_bf16<<<ew((size_t)512 * 480), 256, 0, stream>>>(fc12w, 456, 458, 458, wfc12, 512, 480);
  k_pack_bf16<<<ew((size_t)64 * 1024), 256, 0, stream>>>(fc2w, 16, 1024, 1024, wfc2, 64, 1024);
  k_pack_bf16<<<ew((size_t)64 * 1024), 256, 0, stream>>>(fc3w, 1, 1024, 1024, wfc3, 64, 1024);
  k_pack_biasf<<<ew(640), 256, 0, stream>>>(fc11b, 639, bfc11, 640);
  k_pack_biasf<<<ew(512), 256, 0, stream>>>(fc12b, 456, bfc12, 512);
  k_pack_biasf<<<ew(64), 256, 0, stream>>>(fc2b, 16, bfc2, 64);
  k_pack_biasf<<<ew(64), 256, 0, stream>>>(fc3b, 1, bfc3, 64);
  for (int d = 0; d < 8; ++d) {
    k_pack_gate_w<<<ew((size_t)2048 * dir_kp[d]), 256, 0, stream>>>(
        (const float*)d_in[dirs[d].wih], dir_in[d], HID, wih_pk[d], dir_kp[d]);
    k_pack_gate_w<<<ew((size_t)2048 * 512), 256, 0, stream>>>(
        (const float*)d_in[dirs[d].whh], HID, HID, whh_pk[d], 512);
    k_pack_gate_bias<<<ew(2048), 256, 0, stream>>>(
        (const float*)d_in[dirs[d].bih], (const float*)d_in[dirs[d].bhh], HID, gb[d]);
  }

  auto run_lstm_layer = [&](const bf16_t* X, int Kp, int d0, float* outbuf) {
    int kch = Kp / 32;
    k_wmma_gemm<<<dim3(32, 16), 256, 0, stream>>>(X, Kp, wih_pk[d0],     Kp, gb[d0],     gbufF, 2048, kch, 0);
    k_wmma_gemm<<<dim3(32, 16), 256, 0, stream>>>(X, Kp, wih_pk[d0 + 1], Kp, gb[d0 + 1], gbufB, 2048, kch, 0);
    k_lstm_rec<<<2, 1024, 0, stream>>>(gbufF, gbufB, whh_pk[d0], whh_pk[d0 + 1], outbuf, 1024, T, HID);
  };

  // ---- stack 1 ----
  // X1 = [ling | ac] -> bf16 4096x672 ; fc11 -> relu -> fbuf (ld 640)
  k_build_bf16<<<ew((size_t)T * 672), 256, 0, stream>>>(ling, LING, LING, ac, AC, AC, 0, 0, xbf, T, 672);
  k_wmma_gemm<<<dim3(10, 16), 256, 0, stream>>>(xbf, 672, wfc11, 672, bfc11, fbuf, 640, 21, 1);
  // lstm1 L0 input = [fc11out(639) | labels]
  k_build_bf16<<<ew((size_t)T * 672), 256, 0, stream>>>(fbuf, 639, 640, nullptr, 0, 0, 1, 0, xbf, T, 672);
  run_lstm_layer(xbf, 672, 0, outA);
  // lstm1 L1 input = out1_l0 (1022)
  k_build_bf16<<<ew((size_t)T * 1024), 256, 0, stream>>>(outA, 1022, 1024, nullptr, 0, 0, 0, 0, xbf, T, 1024);
  run_lstm_layer(xbf, 1024, 2, outB);

  // ---- VQ branch ----
  k_gather_fc2<<<ew((size_t)MN * 1024), 256, 0, stream>>>(outB, 1024, mora, MN, HID, T, xbf);
  k_wmma_gemm<<<dim3(1, 2), 256, 0, stream>>>(xbf, 1024, wfc2, 1024, bfc2, zubuf, 64, 32, 0);
  k_vq<<<ew(MN), 256, 0, stream>>>(zubuf, 64, cb, zbuf, out_z, out_zu, MN);
  k_ztmp<<<ew(T), 256, 0, stream>>>(mora, MN, zbuf, ztmp, T);

  // ---- stack 2 ----
  // X2 = [ling | z_tmp] -> fc12 -> relu -> fbuf (ld 512)
  k_build_bf16<<<ew((size_t)T * 480), 256, 0, stream>>>(ling, LING, LING, ztmp, 16, 16, 0, 0, xbf, T, 480);
  k_wmma_gemm<<<dim3(8, 16), 256, 0, stream>>>(xbf, 480, wfc12, 480, bfc12, fbuf, 512, 15, 1);
  // lstm2 L0 input = [fc12out(456) | labels]
  k_build_bf16<<<ew((size_t)T * 480), 256, 0, stream>>>(fbuf, 456, 512, nullptr, 0, 0, 1, 0, xbf, T, 480);
  run_lstm_layer(xbf, 480, 4, outA);
  k_build_bf16<<<ew((size_t)T * 1024), 256, 0, stream>>>(outA, 1022, 1024, nullptr, 0, 0, 0, 0, xbf, T, 1024);
  run_lstm_layer(xbf, 1024, 6, outB);

  // ---- head: fc3 on [relu(h3) | labels] ----
  k_build_bf16<<<ew((size_t)T * 1024), 256, 0, stream>>>(outB, 1022, 1024, nullptr, 0, 0, 1, 1, xbf, T, 1024);
  k_wmma_gemm<<<dim3(1, 16), 256, 0, stream>>>(xbf, 1024, wfc3, 1024, bfc3, fbuf, 64, 32, 0);
  k_copy_y<<<ew(T), 256, 0, stream>>>(fbuf, 64, out_y, T);
}
